// GraphAwareAttention_43911745635098
// MI455X (gfx1250) — matrix operand order repaired and checked by reading, not compiled, not run on hardware
//
#include <hip/hip_runtime.h>

// ---------------- problem constants ----------------
constexpr int kN    = 8192;
constexpr int kD    = 256;
constexpr int kH    = 4;
constexpr int kHD   = 64;
constexpr float kScale = 0.125f;   // 64^-0.5

// ---------------- WMMA types ----------------
typedef __bf16        v16bf __attribute__((ext_vector_type(16)));
typedef float         v8f   __attribute__((ext_vector_type(8)));
typedef unsigned int  u32x4 __attribute__((ext_vector_type(4)));

union FragU { v16bf v; u32x4 q[2]; };

#if defined(__AMDGCN__) && __has_builtin(__builtin_amdgcn_global_load_async_to_lds_b128)
#define HAS_ASYNC_LDS 1
#else
#define HAS_ASYNC_LDS 0
#endif

#if HAS_ASYNC_LDS
// builtin prototype is "vV4i*1V4i*3IiIi": (global int4*, lds int4*, imm offset, imm cpol)
typedef int v4ig __attribute__((vector_size(16)));
typedef __attribute__((address_space(1))) v4ig as1_v4ig;
typedef __attribute__((address_space(3))) v4ig as3_v4ig;
#endif

__device__ __forceinline__ unsigned short f2bf(float f) {
  unsigned int u = __float_as_uint(f);
  u += 0x7FFFu + ((u >> 16) & 1u);          // round-to-nearest-even
  return (unsigned short)(u >> 16);
}

// load a 16-bf16 fragment as two aligned b128 chunks from LDS
__device__ __forceinline__ v16bf frag_ld(const unsigned short* p, int off_lo, int off_hi) {
  FragU f;
  f.q[0] = *(const u32x4*)(p + off_lo);
  f.q[1] = *(const u32x4*)(p + off_hi);
  return f.v;
}

__device__ __forceinline__ v8f wmma_bf16(v16bf a, v16bf b, v8f c) {
  return __builtin_amdgcn_wmma_f32_16x16x32_bf16(false, a, false, b, (short)0, c, false, false);
}

// async 16-byte global -> LDS copy (per-lane addresses), ASYNCcnt-tracked.
__device__ __forceinline__ void async_cp16(const unsigned short* g, unsigned short* l) {
#if HAS_ASYNC_LDS
  as1_v4ig* gp = (as1_v4ig*)(uintptr_t)g;                  // flat == global address
  as3_v4ig* lp = (as3_v4ig*)(unsigned int)(uintptr_t)l;    // flat LDS addr[31:0] = LDS offset
  __builtin_amdgcn_global_load_async_to_lds_b128(gp, lp, 0, 0);
#else
  *(u32x4*)l = *(const u32x4*)g;
#endif
}

__device__ __forceinline__ void async_wait0() {
#if HAS_ASYNC_LDS
#if __has_builtin(__builtin_amdgcn_s_wait_asynccnt)
  __builtin_amdgcn_s_wait_asynccnt(0);
#else
  asm volatile("s_wait_asynccnt 0" ::: "memory");
#endif
#endif
}

// =====================================================================
// Kernel 1: QKV projection.  x[N,256] @ Wqkv[256,768] + b  ->
//   Qb[h][n][d] bf16, Kb[h][n][d] bf16, Vtb[h][d][n] bf16 (V transposed)
// =====================================================================
__global__ __launch_bounds__(128) void qkv_kernel(
    const float* __restrict__ x, const float* __restrict__ Wqkv,
    const float* __restrict__ bqkv,
    unsigned short* __restrict__ Qb, unsigned short* __restrict__ Kb,
    unsigned short* __restrict__ Vtb) {
  __shared__ __align__(16) unsigned short sX[64 * 72];
  __shared__ __align__(16) unsigned short sWt[64 * 72];   // sWt[col][k]

  const int tid = threadIdx.x;
  const int w = tid >> 5, lane = tid & 31;
  const int l16 = lane & 15, hi = (lane >> 4) & 1;
  const int rowblk = blockIdx.x * 64;
  const int colblk = blockIdx.y;            // 0..11
  const int part = colblk >> 2;             // 0=Q 1=K 2=V
  const int head = colblk & 3;

  v8f acc[4] = {};

  for (int kk = 0; kk < 4; ++kk) {
    __syncthreads();
#pragma unroll
    for (int i = 0; i < 32; ++i) {
      int idx = i * 128 + tid;
      int r = idx >> 6, c = idx & 63;
      sX[r * 72 + c] = f2bf(x[(rowblk + r) * kD + kk * 64 + c]);
    }
#pragma unroll
    for (int i = 0; i < 32; ++i) {
      int idx = i * 128 + tid;
      int r = idx >> 6, c = idx & 63;       // r = k, c = col
      sWt[c * 72 + r] = f2bf(Wqkv[(kk * 64 + r) * (3 * kD) + colblk * 64 + c]);
    }
    __syncthreads();
#pragma unroll
    for (int kc = 0; kc < 2; ++kc) {
      v16bf a  = frag_ld(&sX[(w * 16 + l16) * 72],
                         kc * 32 + hi * 8, kc * 32 + 16 + hi * 8);
      v16bf b0 = frag_ld(&sWt[(0 * 16 + l16) * 72], kc * 32 + hi * 8, kc * 32 + 16 + hi * 8);
      v16bf b1 = frag_ld(&sWt[(1 * 16 + l16) * 72], kc * 32 + hi * 8, kc * 32 + 16 + hi * 8);
      v16bf b2 = frag_ld(&sWt[(2 * 16 + l16) * 72], kc * 32 + hi * 8, kc * 32 + 16 + hi * 8);
      v16bf b3 = frag_ld(&sWt[(3 * 16 + l16) * 72], kc * 32 + hi * 8, kc * 32 + 16 + hi * 8);
      acc[0] = wmma_bf16(a, b0, acc[0]);
      acc[1] = wmma_bf16(a, b1, acc[1]);
      acc[2] = wmma_bf16(a, b2, acc[2]);
      acc[3] = wmma_bf16(a, b3, acc[3]);
    }
  }

#pragma unroll
  for (int t = 0; t < 4; ++t) {
    int c = colblk * 64 + t * 16 + l16;
    float bias = bqkv[c];
    int d = t * 16 + l16;
#pragma unroll
    for (int r = 0; r < 8; ++r) {
      int m = rowblk + w * 16 + r + hi * 8;
      unsigned short bv = f2bf(acc[t][r] + bias);
      if (part == 0)      Qb[((size_t)head * kN + m) * kHD + d] = bv;
      else if (part == 1) Kb[((size_t)head * kN + m) * kHD + d] = bv;
      else                Vtb[((size_t)head * kHD + d) * kN + m] = bv;
    }
  }
}

// =====================================================================
// Kernel 2: flash attention with graph bias + mask.
// grid (N/64, H), block 128 (4 waves, 16 query rows per wave).
// Double-buffered K/Vt tiles with async global->LDS copies.
// =====================================================================
__global__ __launch_bounds__(128) void attn_kernel(
    const unsigned short* __restrict__ Qb, const unsigned short* __restrict__ Kb,
    const unsigned short* __restrict__ Vtb,
    const float* __restrict__ adj, const int* __restrict__ mask,
    const float* __restrict__ graph_bias, float* __restrict__ Ob) {
  __shared__ __align__(16) unsigned short sQ[64 * 72];
  __shared__ __align__(16) unsigned short sK[2][64 * 72];
  __shared__ __align__(16) unsigned short sVt[2][64 * 72];
  __shared__ __align__(16) unsigned short sP[64 * 72];

  const int tid = threadIdx.x;
  const int w = tid >> 5, lane = tid & 31;
  const int l16 = lane & 15, hi = (lane >> 4) & 1;
  const int rowblk = blockIdx.x * 64;
  const int h = blockIdx.y;
  const float gb = graph_bias[0];

  const unsigned short* KgBase = Kb + (size_t)h * kN * kHD;   // [key][d]
  const unsigned short* VgBase = Vtb + (size_t)h * kHD * kN;  // [d][key]

  // ---- issue async prefetch of tile 0 into buffer 0 ----
  {
    const unsigned short* Kg = KgBase;          // kb = 0
    const unsigned short* Vg = VgBase;          // + kb = 0
#pragma unroll
    for (int i = 0; i < 4; ++i) {
      int idx = tid * 4 + i;
      int r = idx >> 3, c8 = idx & 7;
      async_cp16(&Kg[r * kHD + c8 * 8], &sK[0][r * 72 + c8 * 8]);
      async_cp16(&Vg[(size_t)r * kN + c8 * 8], &sVt[0][r * 72 + c8 * 8]);
    }
  }

  // ---- Q tile (64 rows x 64 d), loaded once ----
  {
    const unsigned short* Qg = Qb + ((size_t)h * kN + rowblk) * kHD;
#pragma unroll
    for (int i = 0; i < 4; ++i) {
      int idx = tid * 4 + i;
      int r = idx >> 3, c8 = idx & 7;
      *(u32x4*)&sQ[r * 72 + c8 * 8] = *(const u32x4*)&Qg[r * kHD + c8 * 8];
    }
  }
  __syncthreads();

  // Q A-fragments are loop-invariant: hoist out of the key loop.
  v16bf aQ[2];
#pragma unroll
  for (int kc = 0; kc < 2; ++kc)
    aQ[kc] = frag_ld(&sQ[(w * 16 + l16) * 72],
                     kc * 32 + hi * 8, kc * 32 + 16 + hi * 8);

  v8f acc[4] = {};
  float row_max[8], row_sum[8];
#pragma unroll
  for (int r = 0; r < 8; ++r) { row_max[r] = -3.0e38f; row_sum[r] = 0.0f; }

  constexpr int kIters = kN / 64;
  for (int it = 0; it < kIters; ++it) {
    const int cur = it & 1;
    const int kb = it * 64;

    async_wait0();          // this wave's async copies for tile `it` done
    __syncthreads();        // all waves' copies done; prev buffer free

    // issue async prefetch of tile it+1 into the other buffer
    if (it + 1 < kIters) {
      const unsigned short* Kg = KgBase + (size_t)(kb + 64) * kHD;
      const unsigned short* Vg = VgBase + (kb + 64);
#pragma unroll
      for (int i = 0; i < 4; ++i) {
        int idx = tid * 4 + i;
        int r = idx >> 3, c8 = idx & 7;
        async_cp16(&Kg[r * kHD + c8 * 8], &sK[cur ^ 1][r * 72 + c8 * 8]);
        async_cp16(&Vg[(size_t)r * kN + c8 * 8], &sVt[cur ^ 1][r * 72 + c8 * 8]);
      }
    }

    const unsigned short* cK = sK[cur];
    const unsigned short* cV = sVt[cur];

    // ---- S = Q K^T (16x64 per wave) ----
    v8f s[4] = {};
#pragma unroll
    for (int kc = 0; kc < 2; ++kc) {
      v16bf b0 = frag_ld(&cK[(0 * 16 + l16) * 72], kc * 32 + hi * 8, kc * 32 + 16 + hi * 8);
      v16bf b1 = frag_ld(&cK[(1 * 16 + l16) * 72], kc * 32 + hi * 8, kc * 32 + 16 + hi * 8);
      v16bf b2 = frag_ld(&cK[(2 * 16 + l16) * 72], kc * 32 + hi * 8, kc * 32 + 16 + hi * 8);
      v16bf b3 = frag_ld(&cK[(3 * 16 + l16) * 72], kc * 32 + hi * 8, kc * 32 + 16 + hi * 8);
      s[0] = wmma_bf16(aQ[kc], b0, s[0]);
      s[1] = wmma_bf16(aQ[kc], b1, s[1]);
      s[2] = wmma_bf16(aQ[kc], b2, s[2]);
      s[3] = wmma_bf16(aQ[kc], b3, s[3]);
    }

    // ---- scale + graph bias + mask (streamed nontemporal) ----
#pragma unroll
    for (int nt = 0; nt < 4; ++nt) {
      int n = kb + nt * 16 + l16;
#pragma unroll
      for (int r = 0; r < 8; ++r) {
        int m = rowblk + w * 16 + r + hi * 8;
        size_t off = (size_t)m * kN + n;
        float av = __builtin_nontemporal_load(adj + off);
        int   mk = __builtin_nontemporal_load(mask + off);
        float sv = s[nt][r] * kScale + gb * av;
        s[nt][r] = (mk == 0) ? -1.0e9f : sv;
      }
    }

    // ---- online softmax (row lives across 16 lanes in C layout) ----
    float alpha[8];
#pragma unroll
    for (int r = 0; r < 8; ++r) {
      float mx = fmaxf(fmaxf(s[0][r], s[1][r]), fmaxf(s[2][r], s[3][r]));
      mx = fmaxf(mx, __shfl_xor(mx, 1, 16));
      mx = fmaxf(mx, __shfl_xor(mx, 2, 16));
      mx = fmaxf(mx, __shfl_xor(mx, 4, 16));
      mx = fmaxf(mx, __shfl_xor(mx, 8, 16));
      float m_new = fmaxf(row_max[r], mx);
      alpha[r] = __expf(row_max[r] - m_new);
      row_max[r] = m_new;
      float ps = 0.0f;
#pragma unroll
      for (int nt = 0; nt < 4; ++nt) {
        float p = __expf(s[nt][r] - m_new);
        s[nt][r] = p;
        ps += p;
      }
      ps += __shfl_xor(ps, 1, 16);
      ps += __shfl_xor(ps, 2, 16);
      ps += __shfl_xor(ps, 4, 16);
      ps += __shfl_xor(ps, 8, 16);
      row_sum[r] = row_sum[r] * alpha[r] + ps;
    }
#pragma unroll
    for (int t = 0; t < 4; ++t)
#pragma unroll
      for (int r = 0; r < 8; ++r) acc[t][r] = acc[t][r] * alpha[r];

    // ---- P (f32, C layout) -> bf16 A layout via per-wave LDS region ----
#pragma unroll
    for (int nt = 0; nt < 4; ++nt)
#pragma unroll
      for (int r = 0; r < 8; ++r)
        sP[(w * 16 + r + hi * 8) * 72 + nt * 16 + l16] = f2bf(s[nt][r]);
    // same-wave LDS store->load stays in order in the DS pipeline

    // ---- O += P V ----
#pragma unroll
    for (int nc = 0; nc < 2; ++nc) {
      v16bf aP = frag_ld(&sP[(w * 16 + l16) * 72],
                         nc * 32 + hi * 8, nc * 32 + 16 + hi * 8);
      v16bf b0 = frag_ld(&cV[(0 * 16 + l16) * 72], nc * 32 + hi * 8, nc * 32 + 16 + hi * 8);
      v16bf b1 = frag_ld(&cV[(1 * 16 + l16) * 72], nc * 32 + hi * 8, nc * 32 + 16 + hi * 8);
      v16bf b2 = frag_ld(&cV[(2 * 16 + l16) * 72], nc * 32 + hi * 8, nc * 32 + 16 + hi * 8);
      v16bf b3 = frag_ld(&cV[(3 * 16 + l16) * 72], nc * 32 + hi * 8, nc * 32 + 16 + hi * 8);
      acc[0] = wmma_bf16(aP, b0, acc[0]);
      acc[1] = wmma_bf16(aP, b1, acc[1]);
      acc[2] = wmma_bf16(aP, b2, acc[2]);
      acc[3] = wmma_bf16(aP, b3, acc[3]);
    }

    __syncthreads();        // done reading `cur` before it is refilled
  }

  // ---- normalize + write O (merged heads: [n][h*64+d], f32) ----
#pragma unroll
  for (int r = 0; r < 8; ++r) {
    int m = rowblk + w * 16 + r + hi * 8;
    float inv = 1.0f / (row_sum[r] > 0.0f ? row_sum[r] : 1.0f);
#pragma unroll
    for (int t = 0; t < 4; ++t)
      Ob[(size_t)m * kD + h * kHD + t * 16 + l16] = acc[t][r] * inv;
  }
}

// =====================================================================
// Kernel 3: out projection.  O[N,256] @ Wout[256,256] + bout -> out f32
// =====================================================================
__global__ __launch_bounds__(128) void outproj_kernel(
    const float* __restrict__ Ob, const float* __restrict__ Wout,
    const float* __restrict__ bout, float* __restrict__ out) {
  __shared__ __align__(16) unsigned short sX[64 * 72];
  __shared__ __align__(16) unsigned short sWt[64 * 72];

  const int tid = threadIdx.x;
  const int w = tid >> 5, lane = tid & 31;
  const int l16 = lane & 15, hi = (lane >> 4) & 1;
  const int rowblk = blockIdx.x * 64;
  const int colblk = blockIdx.y;            // 0..3

  v8f acc[4] = {};

  for (int kk = 0; kk < 4; ++kk) {
    __syncthreads();
#pragma unroll
    for (int i = 0; i < 32; ++i) {
      int idx = i * 128 + tid;
      int r = idx >> 6, c = idx & 63;
      sX[r * 72 + c] = f2bf(Ob[(size_t)(rowblk + r) * kD + kk * 64 + c]);
    }
#pragma unroll
    for (int i = 0; i < 32; ++i) {
      int idx = i * 128 + tid;
      int r = idx >> 6, c = idx & 63;
      sWt[c * 72 + r] = f2bf(Wout[(kk * 64 + r) * kD + colblk * 64 + c]);
    }
    __syncthreads();
#pragma unroll
    for (int kc = 0; kc < 2; ++kc) {
      v16bf a  = frag_ld(&sX[(w * 16 + l16) * 72],
                         kc * 32 + hi * 8, kc * 32 + 16 + hi * 8);
      v16bf b0 = frag_ld(&sWt[(0 * 16 + l16) * 72], kc * 32 + hi * 8, kc * 32 + 16 + hi * 8);
      v16bf b1 = frag_ld(&sWt[(1 * 16 + l16) * 72], kc * 32 + hi * 8, kc * 32 + 16 + hi * 8);
      v16bf b2 = frag_ld(&sWt[(2 * 16 + l16) * 72], kc * 32 + hi * 8, kc * 32 + 16 + hi * 8);
      v16bf b3 = frag_ld(&sWt[(3 * 16 + l16) * 72], kc * 32 + hi * 8, kc * 32 + 16 + hi * 8);
      acc[0] = wmma_bf16(a, b0, acc[0]);
      acc[1] = wmma_bf16(a, b1, acc[1]);
      acc[2] = wmma_bf16(a, b2, acc[2]);
      acc[3] = wmma_bf16(a, b3, acc[3]);
    }
  }

#pragma unroll
  for (int t = 0; t < 4; ++t) {
    int c = colblk * 64 + t * 16 + l16;
    float bias = bout[c];
#pragma unroll
    for (int r = 0; r < 8; ++r) {
      int m = rowblk + w * 16 + r + hi * 8;
      out[(size_t)m * kD + c] = acc[t][r] + bias;
    }
  }
}

// =====================================================================
extern "C" void kernel_launch(void* const* d_in, const int* in_sizes, int n_in,
                              void* d_out, int out_size, void* d_ws, size_t ws_size,
                              hipStream_t stream) {
  const float* x    = (const float*)d_in[0];
  const float* adj  = (const float*)d_in[1];
  const int*   mask = (const int*)d_in[2];
  const float* Wqkv = (const float*)d_in[3];
  const float* bqkv = (const float*)d_in[4];
  const float* Wout = (const float*)d_in[5];
  const float* bout = (const float*)d_in[6];
  const float* gb   = (const float*)d_in[7];
  float* out = (float*)d_out;

  // workspace: Q, K, Vt bf16 (4 MB each) + O f32 (8 MB) = 20 MB
  unsigned short* Qb  = (unsigned short*)d_ws;
  unsigned short* Kb  = Qb  + (size_t)kH * kN * kHD;
  unsigned short* Vtb = Kb  + (size_t)kH * kN * kHD;
  float*          Ob  = (float*)(Vtb + (size_t)kH * kN * kHD);

  qkv_kernel<<<dim3(kN / 64, 12), 128, 0, stream>>>(x, Wqkv, bqkv, Qb, Kb, Vtb);
  attn_kernel<<<dim3(kN / 64, kH), 128, 0, stream>>>(Qb, Kb, Vtb, adj, mask, gb, Ob);
  outproj_kernel<<<dim3(kN / 64, 4), 128, 0, stream>>>(Ob, Wout, bout, out);
}
